// ConvGraphSelfLoop_19645180412612
// MI455X (gfx1250) — matrix operand order repaired and checked by reading, not compiled, not run on hardware
//
#include <hip/hip_runtime.h>

// ---------------------------------------------------------------------------
// ConvGraphSelfLoop on MI455X (gfx1250):
//   valid  = any(adjacency >= 0) over (E,N)                   [B,V]
//   out    = valid ? relu(features @ kernel + bias) : features [B,V,U]
// GEMM M=65536, K=1024, N=1024 done in bf16 WMMA (f32 accumulate).
// Global->LDS staging uses gfx1250 async-to-LDS (ASYNCcnt) via inline asm.
// ---------------------------------------------------------------------------

typedef __bf16 v16bf __attribute__((ext_vector_type(16)));
typedef __bf16 v8bf  __attribute__((ext_vector_type(8)));
typedef float  v8f   __attribute__((ext_vector_type(8)));
typedef float  v4f   __attribute__((ext_vector_type(4)));

#define USE_ASYNC_ASM 1

#define BM 128
#define BN 128
#define BK 64
#define THREADS 256

union BF16Frag { v16bf v; v8bf h[2]; };

#if USE_ASYNC_ASM
// One 16-byte async global->LDS transfer per lane (tracked by ASYNCcnt).
__device__ __forceinline__ void async_copy16(const void* g, void* l) {
  // generic -> AS(3) addrspacecast strips the LDS aperture: gives LDS byte addr
  unsigned lds_addr =
      (unsigned)(unsigned long long)(__attribute__((address_space(3))) void*)l;
  unsigned long long gaddr = (unsigned long long)g;
  asm volatile("global_load_async_to_lds_b128 %0, %1, off"
               :
               : "v"(lds_addr), "v"(gaddr)
               : "memory");
}
__device__ __forceinline__ void wait_async0() {
  asm volatile("s_wait_asynccnt 0" ::: "memory");
}
#endif

// ---------------------------------------------------------------------------
// 1) validity mask: one wave32 per vertex row of E*N=128 int32 entries
// ---------------------------------------------------------------------------
__global__ void valid_mask_kernel(const int* __restrict__ adj,
                                  int* __restrict__ mask,
                                  int rows, int L) {
  const int gw   = (int)((blockIdx.x * blockDim.x + threadIdx.x) >> 5);
  const int lane = (int)(threadIdx.x & 31u);
  if (gw >= rows) return;
  const int* p = adj + (size_t)gw * (size_t)L;
  bool anyv = false;
  for (int i = lane; i < L; i += 32) anyv |= (p[i] >= 0);
  unsigned long long b = __ballot(anyv);
  if (lane == 0) mask[gw] = (b != 0ull) ? 1 : 0;
}

// ---------------------------------------------------------------------------
// 2) features f32 -> bf16 (row-major [M][F], K contiguous for A fragments)
// ---------------------------------------------------------------------------
__global__ void f32_to_bf16_kernel(const float* __restrict__ in,
                                   __bf16* __restrict__ out,
                                   long long n) {
  long long i = ((long long)blockIdx.x * blockDim.x + threadIdx.x) * 4;
  if (i + 3 < n) {
    v4f v = *(const v4f*)(in + i);
    union { __bf16 h[4]; unsigned long long u; } r;
    r.h[0] = (__bf16)v[0]; r.h[1] = (__bf16)v[1];
    r.h[2] = (__bf16)v[2]; r.h[3] = (__bf16)v[3];
    *(unsigned long long*)(out + i) = r.u;
  }
}

// ---------------------------------------------------------------------------
// 3) weights f32 [F][U] -> bf16 transposed [U][F] (K contiguous for B frags)
// ---------------------------------------------------------------------------
__global__ void wT_bf16_kernel(const float* __restrict__ w,
                               __bf16* __restrict__ wT,
                               int Fdim, int Udim) {
  long long idx = (long long)blockIdx.x * blockDim.x + threadIdx.x;
  if (idx >= (long long)Fdim * Udim) return;
  int f = (int)(idx % Fdim);
  int u = (int)(idx / Fdim);
  wT[idx] = (__bf16)w[(size_t)f * (size_t)Udim + (size_t)u];
}

// ---------------------------------------------------------------------------
// 4) bf16 WMMA GEMM + fused relu/bias/mask epilogue
//    block = 256 threads = 8 wave32, tile 128x128, BK=64
//    waves arranged 2(m) x 4(n): each wave -> 64x32 = 4x2 tiles of 16x16
// ---------------------------------------------------------------------------
__global__ __launch_bounds__(THREADS)
void gemm_bf16_wmma_kernel(const __bf16* __restrict__ A,    // [M][F] bf16
                           const __bf16* __restrict__ Bt,   // [U][F] bf16 (B^T)
                           const float*  __restrict__ bias, // [U]
                           const float*  __restrict__ feat, // [M][F] f32 fallback
                           const int*    __restrict__ mask, // [M]
                           float* __restrict__ out,         // [M][U]
                           int Fdim, int Udim) {
  __shared__ __attribute__((aligned(16))) __bf16 As[BM * BK];
  __shared__ __attribute__((aligned(16))) __bf16 Bs[BN * BK];
  __shared__ int msk[BM];

  const int tid  = (int)threadIdx.x;
  const int lane = tid & 31;
  const int wid  = tid >> 5;
  const int wm   = wid >> 2;   // 0..1 : wave row in block
  const int wn   = wid & 3;    // 0..3 : wave col in block
  const int lr   = lane & 15;  // lane row/col within 16
  const int hi   = lane >> 4;  // half-wave select

  const int mBase = (int)blockIdx.y * BM;
  const int nBase = (int)blockIdx.x * BN;

  if (tid < BM) msk[tid] = mask[mBase + tid];

  const v8f zero = {0.f, 0.f, 0.f, 0.f, 0.f, 0.f, 0.f, 0.f};
  v8f acc[4][2];
#pragma unroll
  for (int i = 0; i < 4; ++i)
#pragma unroll
    for (int j = 0; j < 2; ++j) acc[i][j] = zero;

  for (int k0 = 0; k0 < Fdim; k0 += BK) {
    __syncthreads();  // previous iteration's LDS reads done before overwrite

    const size_t aBase = (size_t)mBase * (size_t)Fdim + (size_t)k0;
    const size_t bBase = (size_t)nBase * (size_t)Fdim + (size_t)k0;
    // 128 rows x 64 bf16 = 1024 16B chunks per tile; 4 chunks per thread.
#pragma unroll
    for (int p = 0; p < (BM * BK) / (8 * THREADS); ++p) {
      const int cid = p * THREADS + tid;
      const int row = cid >> 3;
      const int ko  = (cid & 7) << 3;
#if USE_ASYNC_ASM
      async_copy16(A + aBase + (size_t)row * Fdim + ko, &As[row * BK + ko]);
      async_copy16(Bt + bBase + (size_t)row * Fdim + ko, &Bs[row * BK + ko]);
#else
      *(v8bf*)&As[row * BK + ko] = *(const v8bf*)(A + aBase + (size_t)row * Fdim + ko);
      *(v8bf*)&Bs[row * BK + ko] = *(const v8bf*)(Bt + bBase + (size_t)row * Fdim + ko);
#endif
    }
#if USE_ASYNC_ASM
    wait_async0();
#endif
    __syncthreads();

#pragma unroll
    for (int kk = 0; kk < BK; kk += 32) {
      BF16Frag a[4], b[2];
      // A 16x32 bf16 layout: lanes 0-15 row=lr K 0-7 (v0-3) & 16-23 (v4-7);
      //                      lanes 16-31 row=lr K 8-15       & 24-31.
#pragma unroll
      for (int tm = 0; tm < 4; ++tm) {
        const __bf16* rp = &As[(wm * 64 + tm * 16 + lr) * BK + kk + hi * 8];
        a[tm].h[0] = *(const v8bf*)(rp);
        a[tm].h[1] = *(const v8bf*)(rp + 16);
      }
      // B 32x16 bf16 layout: lanes 0-15 col=lr K 0-7 & 16-23; lanes 16-31 K 8-15 & 24-31.
#pragma unroll
      for (int tn = 0; tn < 2; ++tn) {
        const __bf16* cp = &Bs[(wn * 32 + tn * 16 + lr) * BK + kk + hi * 8];
        b[tn].h[0] = *(const v8bf*)(cp);
        b[tn].h[1] = *(const v8bf*)(cp + 16);
      }
#pragma unroll
      for (int tm = 0; tm < 4; ++tm)
#pragma unroll
        for (int tn = 0; tn < 2; ++tn)
          acc[tm][tn] = __builtin_amdgcn_wmma_f32_16x16x32_bf16(
              false, a[tm].v, false, b[tn].v, (short)0, acc[tm][tn],
              false, false);
    }
  }

  // Epilogue: D layout VGPR r -> row r (+8 for hi lanes), lane -> column.
#pragma unroll
  for (int tn = 0; tn < 2; ++tn) {
    const int col = nBase + wn * 32 + tn * 16 + lr;
    const float bi = bias[col];
#pragma unroll
    for (int tm = 0; tm < 4; ++tm) {
      const int rb = wm * 64 + tm * 16 + hi * 8;
#pragma unroll
      for (int r = 0; r < 8; ++r) {
        const int rowInB = rb + r;
        const size_t grow = (size_t)(mBase + rowInB);
        float v = acc[tm][tn][r] + bi;
        v = v > 0.0f ? v : 0.0f;
        if (!msk[rowInB]) v = feat[grow * (size_t)Fdim + col];  // F==U here
        out[grow * (size_t)Udim + col] = v;
      }
    }
  }
}

// ---------------------------------------------------------------------------
// Host launcher
// ---------------------------------------------------------------------------
extern "C" void kernel_launch(void* const* d_in, const int* in_sizes, int n_in,
                              void* d_out, int out_size, void* d_ws, size_t ws_size,
                              hipStream_t stream) {
  const int*   adj  = (const int*)d_in[0];
  const float* feat = (const float*)d_in[1];
  const float* w    = (const float*)d_in[2];
  const float* bias = (const float*)d_in[3];
  float*       out  = (float*)d_out;

  const int Udim  = in_sizes[3];                 // 1024
  const int Fdim  = in_sizes[2] / Udim;          // 1024
  const int Mrows = in_sizes[1] / Fdim;          // 65536
  const int L     = in_sizes[0] / Mrows;         // 128 = E*N

  // Workspace layout: [mask int32 x M][features bf16 x M*F][weights^T bf16 x U*F]
  char* ws = (char*)d_ws;
  int* mask = (int*)ws;
  size_t off = (((size_t)Mrows * sizeof(int)) + 255) & ~(size_t)255;
  __bf16* wsA = (__bf16*)(ws + off);
  off += (((size_t)Mrows * (size_t)Fdim * 2) + 255) & ~(size_t)255;
  __bf16* wsB = (__bf16*)(ws + off);

  // 1) validity mask (8 wave-rows per 256-thread block)
  valid_mask_kernel<<<dim3((unsigned)((Mrows + 7) / 8)), dim3(256), 0, stream>>>(
      adj, mask, Mrows, L);

  // 2) features -> bf16 (stays resident in 192MB L2 for the GEMM's 8 N-passes)
  const long long nFeat = (long long)Mrows * Fdim;
  f32_to_bf16_kernel<<<dim3((unsigned)((nFeat / 4 + 255) / 256)), dim3(256), 0, stream>>>(
      feat, wsA, nFeat);

  // 3) weights -> bf16, transposed to [U][F]
  const long long nW = (long long)Fdim * Udim;
  wT_bf16_kernel<<<dim3((unsigned)((nW + 255) / 256)), dim3(256), 0, stream>>>(
      w, wsB, Fdim, Udim);

  // 4) WMMA GEMM with fused epilogue
  gemm_bf16_wmma_kernel<<<dim3((unsigned)(Udim / BN), (unsigned)(Mrows / BM)),
                          dim3(THREADS), 0, stream>>>(
      wsA, wsB, bias, feat, mask, out, Fdim, Udim);
}
